// Conv2d_16621523436019
// MI455X (gfx1250) — matrix operand order repaired and checked by reading, not compiled, not run on hardware
//
#include <hip/hip_runtime.h>
#include <hip/hip_bf16.h>
#include <stdint.h>

// MI455X / gfx1250: wave32, WMMA 16x16x32 bf16 -> f32, async global->LDS.
typedef __attribute__((ext_vector_type(16))) __bf16 v16bf;
typedef __attribute__((ext_vector_type(8)))  float  v8f;

#define B_    32
#define CIN_  128
#define COUT_ 256
#define HW_   56
#define NPIX  (HW_*HW_)     // 3136
#define KTOT  (9*CIN_)      // 1152
#define NCHUNK 36           // 9 taps * (128/32) ci-chunks

__device__ __forceinline__ uint16_t f2bf(float f) {
  union { float f; uint32_t u; } v; v.f = f;
  uint32_t u = v.u;
  uint32_t r = u + 0x7FFFu + ((u >> 16) & 1u);   // round-to-nearest-even
  if ((u & 0x7F800000u) == 0x7F800000u) r = u;   // inf/nan: truncate
  return (uint16_t)(r >> 16);
}

// Pass 1: x (NCHW fp32) -> xT (NHWC bf16) and channel-sum xs[b][h][w].
__global__ void prep_kernel(const float* __restrict__ x,
                            uint16_t* __restrict__ xT,
                            float* __restrict__ xs) {
  int idx = blockIdx.x * blockDim.x + threadIdx.x;
  if (idx >= B_ * NPIX) return;
  int b = idx / NPIX, p = idx % NPIX;
  const float* src = x + (size_t)b * CIN_ * NPIX + p;
  uint16_t* dst = xT + (size_t)idx * CIN_;
  float s = 0.f;
#pragma unroll 4
  for (int ci = 0; ci < CIN_; ++ci) {
    float v = src[(size_t)ci * NPIX];
    s += v;
    dst[ci] = f2bf(v);
  }
  xs[idx] = s;
}

// Pass 2: periphery stencil (5x5 minus center 3x3, row-major order) on xs.
__global__ void periph_kernel(const float* __restrict__ xs,
                              const float* __restrict__ pw,
                              float* __restrict__ periph) {
  int idx = blockIdx.x * blockDim.x + threadIdx.x;
  if (idx >= B_ * NPIX) return;
  int b = idx / NPIX, p = idx % NPIX;
  int ho = p / HW_, wo = p % HW_;
  const float* xsb = xs + (size_t)b * NPIX;
  float acc = 0.f; int cnt = 0;
#pragma unroll
  for (int i = 0; i < 5; ++i) {
#pragma unroll
    for (int j = 0; j < 5; ++j) {
      if (i >= 1 && i <= 3 && j >= 1 && j <= 3) continue;  // core region
      float w = pw[cnt++];
      int hi = ho + i - 2, wi = wo + j - 2;
      if (hi >= 0 && hi < HW_ && wi >= 0 && wi < HW_)
        acc += w * xsb[hi * HW_ + wi];
    }
  }
  periph[idx] = acc;
}

// Pass 3: core weights OIHW fp32 -> bf16 Abf[co][(kh*3+kw)*128 + ci].
// Also zeroes the 256-byte OOB zero-page used by the async staging.
__global__ void wconv_kernel(const float* __restrict__ core,
                             uint16_t* __restrict__ Abf,
                             uint16_t* __restrict__ zpad) {
  int idx = blockIdx.x * blockDim.x + threadIdx.x;
  if (blockIdx.x == 0 && threadIdx.x < 128) zpad[threadIdx.x] = 0;
  if (idx >= COUT_ * KTOT) return;
  int co = idx / KTOT, k = idx % KTOT;
  int pos = k / CIN_, ci = k % CIN_;
  Abf[idx] = f2bf(core[((size_t)co * CIN_ + ci) * 9 + pos]);
}

// Issue one async b128 global->LDS copy for chunk c of this block's B panel.
// Each thread moves 8 bf16 (16 B); OOB halo lanes read a zeroed page.
__device__ __forceinline__ void async_stage(const uint16_t* __restrict__ xT,
                                            const uint16_t* __restrict__ zpad,
                                            int b, int ho, int sn, int sseg,
                                            int c, uint32_t ldsdst) {
  int pos = c >> 2, cc = c & 3;
  int kh = pos / 3, kw = pos % 3;
  int hi = ho + kh - 1;
  int wi = sn + kw - 1;
  bool inb = (hi >= 0) && (hi < HW_) && (wi >= 0) && (wi < HW_);
  const uint16_t* src = inb
      ? xT + ((size_t)b * NPIX + (size_t)(hi * HW_ + wi)) * CIN_
           + cc * 32 + sseg * 8
      : zpad + sseg * 8;
  uint64_t ga = (uint64_t)(uintptr_t)src;
  // CDNA5 async DMA into LDS, tracked by ASYNCcnt (CDNA5 ISA)
  asm volatile("global_load_async_to_lds_b128 %0, %1, off"
               :: "v"(ldsdst), "v"(ga)
               : "memory");
}

#define WMMA_BF16(A, Bv, C) \
  __builtin_amdgcn_wmma_f32_16x16x32_bf16(false, (A), false, (Bv), (short)0, (C), false, false)

// Pass 4: implicit-GEMM conv via WMMA bf16, double-buffered async B staging.
// Block: 256 threads = 8 waves; tile = 128 couts x 64 pixels (one (b,ho) row,
// Wo padded 56->64). Wave w: 2 m-tiles at (w%4)*32, 2 n-tiles at (w/4)*32
// -> 4 wmma per K-chunk per wave, 32 per block per barrier pair.
__global__ __launch_bounds__(256)
void conv_wmma_kernel(const uint16_t* __restrict__ xT,
                      const uint16_t* __restrict__ Abf,
                      const uint16_t* __restrict__ zpad,
                      const float* __restrict__ periph,
                      const float* __restrict__ thresh,
                      const float* __restrict__ scale,
                      float* __restrict__ out) {
  __shared__ __align__(16) uint16_t Bs[2][64 * 32];  // [buf][n][k], 2 x 4 KB

  int bid  = blockIdx.x;
  int mblk = bid & 1;            // 2 m-blocks of 128 couts
  int row  = bid >> 1;
  int ho   = row % HW_;
  int b    = row / HW_;

  int tid  = threadIdx.x;
  int wave = tid >> 5;
  int lane = tid & 31;
  int l16  = lane & 15;
  int lh   = lane >> 4;          // K-half select

  int mpair = wave & 3;          // 2 m-tiles at mblk*128 + mpair*32 (+0,+16)
  int n0    = (wave >> 2) * 32;  // 2 n-tiles: n0, n0+16

  // B staging assignment: thread -> (col n = wo, 8-ci segment)
  int sn   = tid & 63;
  int sseg = tid >> 6;

  const uint16_t* Arow0 =
      Abf + (size_t)(mblk * 128 + mpair * 32 + l16) * KTOT;
  const uint16_t* Arow1 = Arow0 + (size_t)16 * KTOT;

  // LDS byte address of this thread's staging slot (low 32 bits of the flat
  // LDS-aperture address are the LDS offset).
  uint32_t mydst = (uint32_t)(uintptr_t)(&Bs[0][0]) + sn * 64 + sseg * 16;

  v8f acc00 = {}, acc01 = {}, acc10 = {}, acc11 = {};

  // Prime the pipeline: chunk 0 -> buffer 0.
  async_stage(xT, zpad, b, ho, sn, sseg, 0, mydst);

#pragma unroll
  for (int c = 0; c < NCHUNK; ++c) {
    int buf = c & 1;
    // Prefetch-issue next panel into the other buffer.
    if (c + 1 < NCHUNK) {
      async_stage(xT, zpad, b, ho, sn, sseg, c + 1,
                  mydst + ((c + 1) & 1) * 4096);
      asm volatile("s_wait_asynccnt 0x1" ::: "memory");  // chunk c landed
    } else {
      asm volatile("s_wait_asynccnt 0x0" ::: "memory");
    }
    __syncthreads();   // all waves' copies for chunk c visible

    int kb = (c >> 2) * 128 + (c & 3) * 32;
    // Keep next A slices warm (emits global_prefetch_b8).
    if (c + 1 < NCHUNK) {
      int kbn = ((c + 1) >> 2) * 128 + ((c + 1) & 3) * 32;
      __builtin_prefetch(Arow0 + kbn, 0, 0);
      __builtin_prefetch(Arow1 + kbn, 0, 0);
    }

    // A fragments (row m = l16, K = lh*16 .. +15): two b128 per lane each.
    union { v16bf v; uint4 q[2]; } a0, a1, b0v, b1v;
    a0.q[0] = *(const uint4*)(Arow0 + kb + lh * 16);
    a0.q[1] = *(const uint4*)(Arow0 + kb + lh * 16 + 8);
    a1.q[0] = *(const uint4*)(Arow1 + kb + lh * 16);
    a1.q[1] = *(const uint4*)(Arow1 + kb + lh * 16 + 8);
    // B fragments (col n = l16, K = lh*16 .. +15): two ds_load_b128 each.
    const uint16_t* Bp = &Bs[buf][0];
    b0v.q[0] = *(const uint4*)(Bp + (n0 + l16) * 32 + lh * 16);
    b0v.q[1] = *(const uint4*)(Bp + (n0 + l16) * 32 + lh * 16 + 8);
    b1v.q[0] = *(const uint4*)(Bp + (n0 + 16 + l16) * 32 + lh * 16);
    b1v.q[1] = *(const uint4*)(Bp + (n0 + 16 + l16) * 32 + lh * 16 + 8);

    acc00 = WMMA_BF16(a0.v, b0v.v, acc00);
    acc01 = WMMA_BF16(a0.v, b1v.v, acc01);
    acc10 = WMMA_BF16(a1.v, b0v.v, acc10);
    acc11 = WMMA_BF16(a1.v, b1v.v, acc11);

    __syncthreads();   // reads done before this buffer is overwritten
  }

  // Epilogue: gate + periphery, fp32 store. C/D layout: M=r+8*lh, N=l16.
  float sc = scale[0];
  int wo0 = n0 + l16;
  int wo1 = n0 + 16 + l16;
  const float* pb = periph + (size_t)b * NPIX + ho * HW_;
  float pv0 = (wo0 < HW_) ? pb[wo0] : 0.f;
  float pv1 = (wo1 < HW_) ? pb[wo1] : 0.f;
  int cobase = mblk * 128 + mpair * 32 + lh * 8;
#pragma unroll
  for (int r = 0; r < 8; ++r) {
    int co0 = cobase + r;          // m-tile 0
    int co1 = cobase + 16 + r;     // m-tile 1
    float th0 = thresh[co0];
    float th1 = thresh[co1];
    float v00 = acc00[r], v01 = acc01[r];
    float v10 = acc10[r], v11 = acc11[r];
    float g00 = 1.f / (1.f + __expf(-sc * (v00 - th0)));
    float g01 = 1.f / (1.f + __expf(-sc * (v01 - th0)));
    float g10 = 1.f / (1.f + __expf(-sc * (v10 - th1)));
    float g11 = 1.f / (1.f + __expf(-sc * (v11 - th1)));
    size_t base0 = ((size_t)b * COUT_ + co0) * NPIX + (size_t)ho * HW_;
    size_t base1 = ((size_t)b * COUT_ + co1) * NPIX + (size_t)ho * HW_;
    if (wo0 < HW_) {
      out[base0 + wo0] = v00 + g00 * pv0;
      out[base1 + wo0] = v10 + g10 * pv0;
    }
    if (wo1 < HW_) {
      out[base0 + wo1] = v01 + g01 * pv1;
      out[base1 + wo1] = v11 + g11 * pv1;
    }
  }
}

extern "C" void kernel_launch(void* const* d_in, const int* in_sizes, int n_in,
                              void* d_out, int out_size, void* d_ws, size_t ws_size,
                              hipStream_t stream) {
  const float* x      = (const float*)d_in[0];
  const float* core   = (const float*)d_in[1];
  const float* pw     = (const float*)d_in[2];
  const float* thresh = (const float*)d_in[3];
  const float* scale  = (const float*)d_in[4];
  float* out = (float*)d_out;

  char* ws = (char*)d_ws;
  uint16_t* xT = (uint16_t*)ws;
  size_t off = (size_t)B_ * NPIX * CIN_ * sizeof(uint16_t);   // 25.7 MB
  float* xs = (float*)(ws + off);      off += (size_t)B_ * NPIX * sizeof(float);
  float* periph = (float*)(ws + off);  off += (size_t)B_ * NPIX * sizeof(float);
  uint16_t* Abf = (uint16_t*)(ws + off);
  off += (size_t)COUT_ * KTOT * sizeof(uint16_t);             // 0.59 MB
  uint16_t* zpad = (uint16_t*)(ws + off);                     // 256 B zero page

  prep_kernel<<<(B_ * NPIX + 255) / 256, 256, 0, stream>>>(x, xT, xs);
  periph_kernel<<<(B_ * NPIX + 255) / 256, 256, 0, stream>>>(xs, pw, periph);
  wconv_kernel<<<(COUT_ * KTOT + 255) / 256, 256, 0, stream>>>(core, Abf, zpad);
  conv_wmma_kernel<<<B_ * HW_ * 2, 256, 0, stream>>>(xT, Abf, zpad, periph,
                                                     thresh, scale, out);
}
